// TwoPathPointNet2_19456201851257
// MI455X (gfx1250) — compile-verified, matrix operand order checked by reading, not run on hardware
//
#include <hip/hip_runtime.h>
#include <hip/hip_bf16.h>
#include <math.h>

typedef __bf16 bf16;
typedef __bf16 v16bf __attribute__((ext_vector_type(16)));
typedef __bf16 v8bf  __attribute__((ext_vector_type(8)));
typedef float  v8f   __attribute__((ext_vector_type(8)));
typedef int    v4i   __attribute__((ext_vector_type(4)));

#define AS1 __attribute__((address_space(1)))
#define AS3 __attribute__((address_space(3)))

#if __has_builtin(__builtin_amdgcn_global_load_async_to_lds_b128) && \
    __has_builtin(__builtin_amdgcn_s_wait_asynccnt)
#define USE_ASYNC_LDS 1
#else
#define USE_ASYNC_LDS 0
#endif

static inline int pad32(int x){ return (x + 31) & ~31; }

__device__ __forceinline__ float gelu_f(float x){
  return 0.5f * x * (1.0f + erff(x * 0.70710678118654752f));
}

// ---------------------------------------------------------------------------
// Layout / gather kernels
// ---------------------------------------------------------------------------
__global__ void k_transpose(const float* __restrict__ x, float* __restrict__ x1,
                            float* __restrict__ x2){
  long i = (long)blockIdx.x * blockDim.x + threadIdx.x;
  long total = (long)8 * 3 * 4096;
  if (i >= total) return;
  int n = (int)(i % 4096); long t = i / 4096; int c = (int)(t % 3); int b = (int)(t / 3);
  x1[((long)b * 4096 + n) * 3 + c] = x[(((long)b * 2 + 0) * 3 + c) * 4096 + n];
  x2[((long)b * 4096 + n) * 3 + c] = x[(((long)b * 2 + 1) * 3 + c) * 4096 + n];
}

__global__ void k_fps(const float* __restrict__ xyz, int N, int S, int* __restrict__ fidx){
  __shared__ float dist[4096];
  __shared__ float sb[256];
  __shared__ int   si[256];
  __shared__ int   sfar;
  int b = blockIdx.x, tid = threadIdx.x;
  for (int i = tid; i < N; i += 256) dist[i] = 1e10f;
  if (tid == 0) sfar = 0;
  __syncthreads();
  for (int it = 0; it < S; ++it){
    int far = sfar;
    if (tid == 0) fidx[b * S + it] = far;
    float cx = xyz[((long)b * N + far) * 3 + 0];
    float cy = xyz[((long)b * N + far) * 3 + 1];
    float cz = xyz[((long)b * N + far) * 3 + 2];
    float best = -1.0f; int bi = 0;
    for (int i = tid; i < N; i += 256){
      float dx = xyz[((long)b * N + i) * 3 + 0] - cx;
      float dy = xyz[((long)b * N + i) * 3 + 1] - cy;
      float dz = xyz[((long)b * N + i) * 3 + 2] - cz;
      float d = dx * dx + dy * dy + dz * dz;
      float nd = fminf(dist[i], d);
      dist[i] = nd;
      if (nd > best){ best = nd; bi = i; }
    }
    sb[tid] = best; si[tid] = bi;
    __syncthreads();
    for (int s = 128; s > 0; s >>= 1){
      if (tid < s){
        if (sb[tid + s] > sb[tid] || (sb[tid + s] == sb[tid] && si[tid + s] < si[tid])){
          sb[tid] = sb[tid + s]; si[tid] = si[tid + s];
        }
      }
      __syncthreads();
    }
    if (tid == 0) sfar = si[0];
    __syncthreads();
  }
}

__global__ void k_gather3(const float* __restrict__ xyz, const int* __restrict__ fidx,
                          float* __restrict__ nxyz, int B, int N, int S){
  int i = blockIdx.x * blockDim.x + threadIdx.x;
  if (i >= B * S) return;
  int b = i / S; int n = fidx[i];
  for (int d = 0; d < 3; ++d) nxyz[(long)i * 3 + d] = xyz[((long)b * N + n) * 3 + d];
}

// wave-per-query ball query: first K neighbors in index order (matches jnp.sort path)
__global__ void k_ballquery(const float* __restrict__ xyz, const float* __restrict__ nxyz,
                            int* __restrict__ gidx, int B, int N, int S, int K, float r2){
  int gw = (blockIdx.x * blockDim.x + threadIdx.x) >> 5;
  int lane = threadIdx.x & 31;
  if (gw >= B * S) return;
  int b = gw / S;
  float cx = nxyz[(long)gw * 3 + 0];
  float cy = nxyz[(long)gw * 3 + 1];
  float cz = nxyz[(long)gw * 3 + 2];
  int* out = gidx + (long)gw * K;
  int cnt = 0; int firstIdx = 0; bool haveFirst = false;
  for (int base = 0; base < N && cnt < K; base += 32){
    int n = base + lane;
    bool in = false;
    if (n < N){
      float dx = xyz[((long)b * N + n) * 3 + 0] - cx;
      float dy = xyz[((long)b * N + n) * 3 + 1] - cy;
      float dz = xyz[((long)b * N + n) * 3 + 2] - cz;
      in = (dx * dx + dy * dy + dz * dz) <= r2;
    }
    unsigned mm = (unsigned)__ballot(in);
    if (!haveFirst && mm){ firstIdx = base + __ffs(mm) - 1; haveFirst = true; }
    if (in){
      int p = cnt + __popc(mm & ((1u << lane) - 1u));
      if (p < K) out[p] = n;
    }
    cnt += __popc(mm);
  }
  if (cnt > K) cnt = K;
  for (int p = cnt + lane; p < K; p += 32) out[p] = firstIdx;
}

// grouped rows: [points(gidx) | xyz(gidx)-center | 0-pad] -> bf16, row stride Kp
__global__ void k_group(const float* __restrict__ xyz, const float* __restrict__ nxyz,
                        const bf16* __restrict__ pts, int Cpts,
                        const int* __restrict__ gidx, bf16* __restrict__ out,
                        int B, int N, int S, int K, int Kp){
  long row = blockIdx.x;            // 0 .. B*S*K-1
  int tid = threadIdx.x;            // 64
  long bs = row / K; int k = (int)(row % K);
  int b = (int)(bs / S);
  int n = gidx[bs * K + k];
  const float* c = nxyz + bs * 3;
  bf16* o = out + row * Kp;
  for (int ch = tid; ch < Kp; ch += 64){
    float v;
    if (ch < Cpts)            v = (float)pts[((long)b * N + n) * Cpts + ch];
    else if (ch < Cpts + 3)   v = xyz[((long)b * N + n) * 3 + (ch - Cpts)] - c[ch - Cpts];
    else                      v = 0.0f;
    o[ch] = (bf16)v;
  }
}

// group_all rows: [xyz | points | 0-pad]
__global__ void k_groupall(const float* __restrict__ xyz, const bf16* __restrict__ pts,
                           int Cpts, bf16* __restrict__ out, int B, int K, int Kp){
  long row = blockIdx.x;            // 0 .. B*K-1
  int tid = threadIdx.x;
  bf16* o = out + row * Kp;
  for (int ch = tid; ch < Kp; ch += 64){
    float v;
    if (ch < 3)               v = xyz[row * 3 + ch];
    else if (ch < 3 + Cpts)   v = (float)pts[row * (long)Cpts + (ch - 3)];
    else                      v = 0.0f;
    o[ch] = (bf16)v;
  }
}

// ---------------------------------------------------------------------------
// WMMA bf16 GEMM:  C(MxN) = A(MxK) * B   (B stored [N][K] if btrans, else [K][N])
// 256-thread block = 8 waves; block tile 128x64; wave tile 16x64 (4 WMMAs/K-step)
// A/B K-panels staged in LDS (async global->LDS DMA when available)
// ---------------------------------------------------------------------------
__global__ void __launch_bounds__(256)
k_gemm(const bf16* __restrict__ A, const bf16* __restrict__ Bm,
       float* __restrict__ C,
       int M, int N, int Kt, int lda, int ldb, int ldc, int H,
       long sAb, long sAh, long sBb, long sBh, long sCb, long sCh,
       int btrans){
  __shared__ bf16 As[128][40];   // 32 used, padded to 40 to spread LDS banks
  __shared__ bf16 Bs[64][40];
  int z = blockIdx.z; int zb = z / H, zh = z % H;
  const bf16* Ab = A + zb * sAb + zh * sAh;
  const bf16* Bb = Bm + zb * sBb + zh * sBh;
  float* Cb = C + zb * sCb + zh * sCh;
  int m0 = blockIdx.x * 128, n0 = blockIdx.y * 64;
  int tid = threadIdx.x;
  int wave = tid >> 5, lane = tid & 31;
  int lr = lane & 15, half = lane >> 4;

  v8f acc[4];
#pragma unroll
  for (int t = 0; t < 4; ++t) acc[t] = (v8f){0.f,0.f,0.f,0.f,0.f,0.f,0.f,0.f};

  for (int k = 0; k < Kt; k += 32){
    // ---- stage A panel: 128 rows x 32 (16B chunks, 2 per thread) ----
#pragma unroll
    for (int i = 0; i < 2; ++i){
      int chunk = tid + i * 256;           // 0..511
      int row = chunk >> 2, coff = (chunk & 3) * 8;
      int grow = m0 + row; if (grow > M - 1) grow = M - 1;
      const bf16* gp = Ab + (long)grow * lda + k + coff;
      bf16* lp = &As[row][coff];
#if USE_ASYNC_LDS
      __builtin_amdgcn_global_load_async_to_lds_b128((AS1 v4i*)gp, (AS3 v4i*)lp, 0, 0);
#else
      *(uint4*)lp = *(const uint4*)gp;
#endif
    }
    // ---- stage B panel: 64 cols x 32 ----
    if (btrans){
      int chunk = tid;                     // 0..255 = 64 rows x 4 chunks
      int row = chunk >> 2, coff = (chunk & 3) * 8;
      int gcol = n0 + row; if (gcol > N - 1) gcol = N - 1;
      const bf16* gp = Bb + (long)gcol * ldb + k + coff;
      bf16* lp = &Bs[row][coff];
#if USE_ASYNC_LDS
      __builtin_amdgcn_global_load_async_to_lds_b128((AS1 v4i*)gp, (AS3 v4i*)lp, 0, 0);
#else
      *(uint4*)lp = *(const uint4*)gp;
#endif
    } else {
#pragma unroll
      for (int i = 0; i < 8; ++i){
        int idx = tid * 8 + i;             // 0..2047
        int nn = idx & 63, kk = idx >> 6;
        int gcol = n0 + nn; if (gcol > N - 1) gcol = N - 1;
        Bs[nn][kk] = Bb[(long)(k + kk) * ldb + gcol];
      }
    }
#if USE_ASYNC_LDS
    __builtin_amdgcn_s_wait_asynccnt(0);
#endif
    __syncthreads();

    // ---- compute: one A fragment reused across 4 B fragments ----
    int ar = wave * 16 + lr;
    v16bf a;
    {
      v8bf lo = *(const v8bf*)&As[ar][half * 8];
      v8bf hi = *(const v8bf*)&As[ar][half * 8 + 16];
#pragma unroll
      for (int e = 0; e < 8; ++e){ a[e] = lo[e]; a[8 + e] = hi[e]; }
    }
#pragma unroll
    for (int t = 0; t < 4; ++t){
      int bc = t * 16 + lr;
      v8bf lo = *(const v8bf*)&Bs[bc][half * 8];
      v8bf hi = *(const v8bf*)&Bs[bc][half * 8 + 16];
      v16bf b;
#pragma unroll
      for (int e = 0; e < 8; ++e){ b[e] = lo[e]; b[8 + e] = hi[e]; }
      acc[t] = __builtin_amdgcn_wmma_f32_16x16x32_bf16(false, a, false, b,
                                                       (short)0, acc[t], false, false);
    }
    __syncthreads();
  }

#pragma unroll
  for (int t = 0; t < 4; ++t){
#pragma unroll
    for (int v = 0; v < 8; ++v){
      int r = m0 + wave * 16 + v + half * 8;
      int c = n0 + t * 16 + lr;
      if (r < M && c < N) Cb[(long)r * ldc + c] = acc[t][v];
    }
  }
}

// ---------------------------------------------------------------------------
// Post-GEMM kernels
// ---------------------------------------------------------------------------
__global__ void k_pack(const float* __restrict__ W, bf16* __restrict__ out,
                       int Co, int Cin, int Kp){
  long i = (long)blockIdx.x * blockDim.x + threadIdx.x;
  if (i >= (long)Co * Kp) return;
  int c = (int)(i % Kp); long r = i / Kp;
  out[i] = (c < Cin) ? (bf16)W[r * Cin + c] : (bf16)0.0f;
}

__global__ void k_colstats(const float* __restrict__ Y, long M, int C,
                           float* __restrict__ st, int rowsPer){
  int c = blockIdx.x * 64 + threadIdx.x;
  if (c >= C) return;
  long r0 = (long)blockIdx.y * rowsPer;
  long r1 = r0 + rowsPer; if (r1 > M) r1 = M;
  float s = 0.f, q = 0.f;
  for (long r = r0; r < r1; ++r){ float v = Y[r * C + c]; s += v; q += v * v; }
  atomicAdd(&st[c], s);
  atomicAdd(&st[C + c], q);
}

__global__ void k_bngelu(const float* __restrict__ Y, const float* __restrict__ st,
                         const float* __restrict__ g, const float* __restrict__ be,
                         long M, int C, bf16* __restrict__ out){
  long i = (long)blockIdx.x * blockDim.x + threadIdx.x;
  if (i >= M * C) return;
  int c = (int)(i % C);
  float mean = st[c] / (float)M;
  float var = st[C + c] / (float)M - mean * mean;
  var = fmaxf(var, 0.0f);
  float x = (Y[i] - mean) * rsqrtf(var + 1e-5f) * g[c] + be[c];
  out[i] = (bf16)gelu_f(x);
}

__global__ void k_biasbf(const float* __restrict__ Y, const float* __restrict__ bias,
                         bf16* __restrict__ out, long n, int C){
  long i = (long)blockIdx.x * blockDim.x + threadIdx.x;
  if (i >= n) return;
  float v = Y[i];
  if (bias) v += bias[i % C];
  out[i] = (bf16)v;
}

__global__ void k_maxpool(const bf16* __restrict__ act, int B, int S, int K, int C,
                          bf16* __restrict__ out, int outStride, int chOff){
  long i = (long)blockIdx.x * blockDim.x + threadIdx.x;
  if (i >= (long)B * S * C) return;
  int c = (int)(i % C); long bs = i / C;
  const bf16* p = act + bs * (long)K * C + c;
  float m = -3.4e38f;
  for (int k = 0; k < K; ++k) m = fmaxf(m, (float)p[(long)k * C]);
  out[bs * outStride + chOff + c] = (bf16)m;
}

__global__ void k_pool(const bf16* __restrict__ feat, int B, int S, int C,
                       bf16* __restrict__ out, int outStride, int maxOff, int meanOff){
  long i = (long)blockIdx.x * blockDim.x + threadIdx.x;
  if (i >= (long)B * C) return;
  int c = (int)(i % C); int b = (int)(i / C);
  const bf16* p = feat + (long)b * S * C + c;
  float mx = -3.4e38f, sm = 0.f;
  for (int s = 0; s < S; ++s){ float v = (float)p[(long)s * C]; mx = fmaxf(mx, v); sm += v; }
  out[(long)b * outStride + maxOff + c]  = (bf16)mx;
  out[(long)b * outStride + meanOff + c] = (bf16)(sm / (float)S);
}

__global__ void k_softmax(const float* __restrict__ S_, bf16* __restrict__ out,
                          long rows, int L, float scale){
  long r = (long)blockIdx.x * (blockDim.x >> 5) + (threadIdx.x >> 5);
  int lane = threadIdx.x & 31;
  if (r >= rows) return;
  const float* p = S_ + r * (long)L;
  bf16* o = out + r * (long)L;
  float mx = -3.4e38f;
  for (int i = lane; i < L; i += 32) mx = fmaxf(mx, p[i] * scale);
  for (int off = 16; off; off >>= 1) mx = fmaxf(mx, __shfl_xor(mx, off, 32));
  float sum = 0.f;
  for (int i = lane; i < L; i += 32) sum += __expf(p[i] * scale - mx);
  for (int off = 16; off; off >>= 1) sum += __shfl_xor(sum, off, 32);
  float inv = 1.0f / sum;
  for (int i = lane; i < L; i += 32) o[i] = (bf16)(__expf(p[i] * scale - mx) * inv);
}

__global__ void k_decnorm(const float* __restrict__ Y, const float* __restrict__ g,
                          const float* __restrict__ be, bf16* __restrict__ outb,
                          int outStride, int outOff, float* __restrict__ outf, int Bn){
  int c = threadIdx.x;           // 512 columns
  float mean = 0.f;
  for (int r = 0; r < Bn; ++r) mean += Y[r * 512 + c];
  mean /= (float)Bn;
  float var = 0.f;
  for (int r = 0; r < Bn; ++r){ float d = Y[r * 512 + c] - mean; var += d * d; }
  var /= (float)Bn;
  float s = rsqrtf(var + 1e-5f);
  for (int r = 0; r < Bn; ++r){
    float v = gelu_f((Y[r * 512 + c] - mean) * s * g[c] + be[c]);
    if (outb) outb[(long)r * outStride + outOff + c] = (bf16)v;
    if (outf) outf[r * 512 + c] = v;
  }
}

// ---------------------------------------------------------------------------
// Host side
// ---------------------------------------------------------------------------
static void gemm(hipStream_t st, const bf16* A, const bf16* B, float* C,
                 int M, int N, int Kt, int lda, int ldb, int ldc,
                 int batches, int H, long sAb, long sAh, long sBb, long sBh,
                 long sCb, long sCh, int btrans){
  dim3 g((M + 127) / 128, (N + 63) / 64, batches);
  k_gemm<<<g, 256, 0, st>>>(A, B, C, M, N, Kt, lda, ldb, ldc, H,
                            sAb, sAh, sBb, sBh, sCb, sCh, btrans);
}
static void gemm1(hipStream_t st, const bf16* A, const bf16* B, float* C,
                  int M, int N, int Kt, int lda, int ldb, int ldc){
  gemm(st, A, B, C, M, N, Kt, lda, ldb, ldc, 1, 1, 0, 0, 0, 0, 0, 0, 1);
}

struct PConv { const bf16* Wp; const float* g; const float* be; int co; int kin; int cin; };
struct AttnP {
  const float *Wk,*Wo,*Wq,*Wv,*bk,*bo,*bq,*bv;
  const bf16 *Wqp,*Wkp,*Wvp,*Wop;
  int F, H;
};

extern "C" void kernel_launch(void* const* d_in, const int* in_sizes, int n_in,
                              void* d_out, int out_size, void* d_ws, size_t ws_size,
                              hipStream_t stream) {
  (void)in_sizes; (void)n_in; (void)out_size; (void)ws_size;
  const int B = 8;

  // -------- parameter parsing (pytree sorted-key order; x first) --------
  const float* x = (const float*)d_in[0];
  int pi = 1;
  auto P = [&]() -> const float* { return (const float*)d_in[pi++]; };

  AttnP ca[3];
  int Fs[3] = {320, 640, 1024}; int Hs[3] = {10, 20, 32};
  for (int i = 0; i < 3; ++i){
    ca[i].Wk = P(); ca[i].Wo = P(); ca[i].Wq = P(); ca[i].Wv = P();
    ca[i].bk = P(); ca[i].bo = P(); ca[i].bq = P(); ca[i].bv = P();
    ca[i].F = Fs[i]; ca[i].H = Hs[i];
  }
  const float *decW[3], *decG[3], *decBe[3];
  int fins[3] = {4096, 3072, 1792};
  for (int i = 0; i < 3; ++i){ decW[i] = P(); decG[i] = P(); decBe[i] = P(); }

  const float* rsa1[2][3][3][4];
  const float* rsa2[2][3][3][4];
  const float* rsa3[2][3][4];
  for (int p = 0; p < 2; ++p){
    for (int br = 0; br < 3; ++br) for (int l = 0; l < 3; ++l) for (int t = 0; t < 4; ++t) rsa1[p][br][l][t] = P();
    for (int br = 0; br < 3; ++br) for (int l = 0; l < 3; ++l) for (int t = 0; t < 4; ++t) rsa2[p][br][l][t] = P();
    for (int l = 0; l < 3; ++l) for (int t = 0; t < 4; ++t) rsa3[p][l][t] = P();
  }

  int MLP1d[3][3] = {{32,32,64},{64,64,128},{64,96,128}};
  int MLP2d[3][3] = {{64,64,128},{128,128,256},{128,128,256}};
  int MLP3d[3]    = {256,512,1024};
  int   K1a[3] = {16,32,128};  float R1a[3] = {0.1f,0.2f,0.4f};
  int   K2a[3] = {32,64,128};  float R2a[3] = {0.2f,0.4f,0.8f};

  // -------- workspace carving (deterministic bump allocator) --------
  char* base = (char*)d_ws; size_t woff = 0;
  auto alloc = [&](size_t bytes) -> void* {
    void* p = base + woff; woff += (bytes + 255) & ~(size_t)255; return p;
  };
  float* xyzP[2]; xyzP[0] = (float*)alloc((size_t)B*4096*3*4); xyzP[1] = (float*)alloc((size_t)B*4096*3*4);
  int*   fidx = (int*)alloc((size_t)B*512*4);
  int*   gidx = (int*)alloc((size_t)B*512*128*4);
  float* nx1[2]; nx1[0] = (float*)alloc((size_t)B*512*3*4); nx1[1] = (float*)alloc((size_t)B*512*3*4);
  float* nx2[2]; nx2[0] = (float*)alloc((size_t)B*128*3*4); nx2[1] = (float*)alloc((size_t)B*128*3*4);
  bf16* f1[2]; f1[0] = (bf16*)alloc((size_t)B*512*320*2); f1[1] = (bf16*)alloc((size_t)B*512*320*2);
  bf16* f2[2]; f2[0] = (bf16*)alloc((size_t)B*128*640*2); f2[1] = (bf16*)alloc((size_t)B*128*640*2);
  bf16* f3[2]; f3[0] = (bf16*)alloc((size_t)B*1024*2);    f3[1] = (bf16*)alloc((size_t)B*1024*2);
  bf16* cf1 = (bf16*)alloc((size_t)B*512*320*2);
  bf16* cf2 = (bf16*)alloc((size_t)B*128*640*2);
  bf16* cf3 = (bf16*)alloc((size_t)B*1024*2);
  bf16*  Gbuf  = (bf16*) alloc((size_t)131072*352*2);      // largest grouped tensor (SA2 br3)
  float* Ybuf  = (float*)alloc((size_t)524288*128*4);      // GEMM f32 out / attention scores
  float* Ybuf2 = (float*)alloc((size_t)8*512*640*4);       // small f32 GEMM outs
  bf16*  ActA  = (bf16*) alloc((size_t)524288*128*2);
  bf16*  ActB  = (bf16*) alloc((size_t)524288*128*2);
  bf16*  Qbf = (bf16*)alloc((size_t)B*512*320*2);
  bf16*  Kbf = (bf16*)alloc((size_t)B*512*320*2);
  bf16*  Vbf = (bf16*)alloc((size_t)B*512*320*2);
  bf16*  attnbf = (bf16*)alloc((size_t)80*512*512*2);
  bf16*  abuf   = (bf16*)alloc((size_t)B*512*320*2);
  float* stats  = (float*)alloc((size_t)2*1024*4);
  bf16* decin1 = (bf16*)alloc((size_t)B*4096*2);
  bf16* decin2 = (bf16*)alloc((size_t)B*3072*2);
  bf16* decin3 = (bf16*)alloc((size_t)B*1792*2);
  bf16* decdum = (bf16*)alloc((size_t)B*512*2);

  // -------- weight packing (f32 -> padded bf16) --------
  auto packW = [&](const float* W, int Co, int Cin, int Kp) -> const bf16* {
    bf16* p = (bf16*)alloc((size_t)Co * Kp * 2);
    long n = (long)Co * Kp;
    k_pack<<<(unsigned)((n + 255) / 256), 256, 0, stream>>>(W, p, Co, Cin, Kp);
    return p;
  };

  PConv c1[2][3][3], c2[2][3][3], c3[2][3];
  for (int p = 0; p < 2; ++p){
    for (int br = 0; br < 3; ++br){
      int cin = 3, kin = pad32(3);
      for (int l = 0; l < 3; ++l){
        int co = MLP1d[br][l];
        c1[p][br][l].Wp = packW(rsa1[p][br][l][0], co, cin, kin);
        c1[p][br][l].g = rsa1[p][br][l][2]; c1[p][br][l].be = rsa1[p][br][l][3];
        c1[p][br][l].co = co; c1[p][br][l].kin = kin; c1[p][br][l].cin = cin;
        cin = co; kin = co;
      }
    }
    for (int br = 0; br < 3; ++br){
      int cin = 323, kin = pad32(323);
      for (int l = 0; l < 3; ++l){
        int co = MLP2d[br][l];
        c2[p][br][l].Wp = packW(rsa2[p][br][l][0], co, cin, kin);
        c2[p][br][l].g = rsa2[p][br][l][2]; c2[p][br][l].be = rsa2[p][br][l][3];
        c2[p][br][l].co = co; c2[p][br][l].kin = kin; c2[p][br][l].cin = cin;
        cin = co; kin = co;
      }
    }
    int cin = 643, kin = pad32(643);
    for (int l = 0; l < 3; ++l){
      int co = MLP3d[l];
      c3[p][l].Wp = packW(rsa3[p][l][0], co, cin, kin);
      c3[p][l].g = rsa3[p][l][2]; c3[p][l].be = rsa3[p][l][3];
      c3[p][l].co = co; c3[p][l].kin = kin; c3[p][l].cin = cin;
      cin = co; kin = co;
    }
  }
  for (int i = 0; i < 3; ++i){
    int F = ca[i].F;
    ca[i].Wqp = packW(ca[i].Wq, F, F, F);
    ca[i].Wkp = packW(ca[i].Wk, F, F, F);
    ca[i].Wvp = packW(ca[i].Wv, F, F, F);
    ca[i].Wop = packW(ca[i].Wo, F, F, F);
  }
  const bf16* decWp[3];
  for (int i = 0; i < 3; ++i) decWp[i] = packW(decW[i], 512, fins[i], fins[i]);

  // -------- fused helpers --------
  auto conv_bn = [&](const bf16* act, long M, const PConv& c, bf16* outAct){
    gemm1(stream, act, c.Wp, Ybuf, (int)M, c.co, c.kin, c.kin, c.kin, c.co);
    (void)hipMemsetAsync(stats, 0, (size_t)2 * c.co * 4, stream);
    int rowsPer = 4096;
    dim3 g((c.co + 63) / 64, (unsigned)((M + rowsPer - 1) / rowsPer));
    k_colstats<<<g, 64, 0, stream>>>(Ybuf, M, c.co, stats, rowsPer);
    long n = M * c.co;
    k_bngelu<<<(unsigned)((n + 255) / 256), 256, 0, stream>>>(Ybuf, stats, c.g, c.be, M, c.co, outAct);
  };

  auto run_sa = [&](const float* xyz, int N, const bf16* pts, int Cpts, int S,
                    const int* Ks, const float* Rs, PConv (&convs)[3][3],
                    float* nxyz_out, bf16* fout, int Ctot){
    k_fps<<<B, 256, 0, stream>>>(xyz, N, S, fidx);
    k_gather3<<<(B * S + 255) / 256, 256, 0, stream>>>(xyz, fidx, nxyz_out, B, N, S);
    int chOff = 0;
    for (int br = 0; br < 3; ++br){
      int K = Ks[br]; float r = Rs[br];
      int nwav = B * S;
      k_ballquery<<<(nwav * 32 + 255) / 256, 256, 0, stream>>>(xyz, nxyz_out, gidx, B, N, S, K, r * r);
      int Kp = pad32(Cpts + 3);
      long M = (long)B * S * K;
      k_group<<<(unsigned)M, 64, 0, stream>>>(xyz, nxyz_out, pts, Cpts, gidx, Gbuf, B, N, S, K, Kp);
      const bf16* cur = Gbuf; bf16* outs[2] = {ActA, ActB};
      int co = 0;
      for (int l = 0; l < 3; ++l){
        conv_bn(cur, M, convs[br][l], outs[l & 1]);
        cur = outs[l & 1]; co = convs[br][l].co;
      }
      long n = (long)B * S * co;
      k_maxpool<<<(unsigned)((n + 255) / 256), 256, 0, stream>>>(cur, B, S, K, co, fout, Ctot, chOff);
      chOff += co;
    }
  };

  auto run_attn = [&](AttnP& a, const bf16* q, const bf16* kv, int Sq, int Sk, bf16* cf){
    int F = a.F, H = a.H, hd = F / H;
    long Mq = (long)B * Sq, Mk = (long)B * Sk;
    gemm1(stream, q, a.Wqp, Ybuf2, (int)Mq, F, F, F, F, F);
    k_biasbf<<<(unsigned)((Mq * F + 255) / 256), 256, 0, stream>>>(Ybuf2, a.bq, Qbf, Mq * F, F);
    gemm1(stream, kv, a.Wkp, Ybuf2, (int)Mk, F, F, F, F, F);
    k_biasbf<<<(unsigned)((Mk * F + 255) / 256), 256, 0, stream>>>(Ybuf2, a.bk, Kbf, Mk * F, F);
    gemm1(stream, kv, a.Wvp, Ybuf2, (int)Mk, F, F, F, F, F);
    k_biasbf<<<(unsigned)((Mk * F + 255) / 256), 256, 0, stream>>>(Ybuf2, a.bv, Vbf, Mk * F, F);
    const bf16* attin;
    if (Sk > 1){
      // scores[b,h,q,k] = Q . K
      gemm(stream, Qbf, Kbf, Ybuf, Sq, Sk, hd, F, F, Sk, B * H, H,
           (long)Sq * F, (long)hd, (long)Sk * F, (long)hd,
           (long)H * Sq * Sk, (long)Sq * Sk, 1);
      long rows = (long)B * H * Sq;
      k_softmax<<<(unsigned)((rows + 7) / 8), 256, 0, stream>>>(Ybuf, attnbf, rows, Sk,
                                                               1.0f / sqrtf((float)F));
      // out[b,q,h*hd+d] = attn . V
      gemm(stream, attnbf, Vbf, Ybuf2, Sq, hd, Sk, Sk, F, F, B * H, H,
           (long)H * Sq * Sk, (long)Sq * Sk, (long)Sk * F, (long)hd,
           (long)Sq * F, (long)hd, 0);
      k_biasbf<<<(unsigned)((Mq * F + 255) / 256), 256, 0, stream>>>(Ybuf2, (const float*)nullptr,
                                                                    abuf, Mq * F, F);
      attin = abuf;
    } else {
      attin = Vbf;  // softmax over 1 element == 1 -> attn output is exactly V
    }
    gemm1(stream, attin, a.Wop, Ybuf2, (int)Mq, F, F, F, F, F);
    k_biasbf<<<(unsigned)((Mq * F + 255) / 256), 256, 0, stream>>>(Ybuf2, a.bo, cf, Mq * F, F);
  };

  // -------- forward pass --------
  {
    long tot = (long)B * 3 * 4096;
    k_transpose<<<(unsigned)((tot + 255) / 256), 256, 0, stream>>>(x, xyzP[0], xyzP[1]);
  }
  for (int p = 0; p < 2; ++p)
    run_sa(xyzP[p], 4096, nullptr, 0, 512, K1a, R1a, c1[p], nx1[p], f1[p], 320);
  run_attn(ca[0], f1[0], f1[1], 512, 512, cf1);

  for (int p = 0; p < 2; ++p)
    run_sa(nx1[p], 512, f1[p], 320, 128, K2a, R2a, c2[p], nx2[p], f2[p], 640);
  run_attn(ca[1], f2[0], f2[1], 128, 128, cf2);

  // SA3 group-all
  for (int p = 0; p < 2; ++p){
    long M = (long)B * 128; int Kp = pad32(643);
    k_groupall<<<(unsigned)M, 64, 0, stream>>>(nx2[p], f2[p], 640, Gbuf, B, 128, Kp);
    const bf16* cur = Gbuf; bf16* outs[2] = {ActA, ActB};
    int co = 0;
    for (int l = 0; l < 3; ++l){ conv_bn(cur, M, c3[p][l], outs[l & 1]); cur = outs[l & 1]; co = c3[p][l].co; }
    long n = (long)B * 1 * co;
    k_maxpool<<<(unsigned)((n + 255) / 256), 256, 0, stream>>>(cur, B, 1, 128, co, f3[p], co, 0);
  }
  run_attn(ca[2], f3[0], f3[1], 1, 1, cf3);

  // -------- decoder --------
  // dec1 input: pool(concat[p1f3, cf3]) -> (B,4096)
  k_pool<<<(B * 1024 + 255) / 256, 256, 0, stream>>>(f3[0], B, 1, 1024, decin1, 4096, 0, 2048);
  k_pool<<<(B * 1024 + 255) / 256, 256, 0, stream>>>(cf3,   B, 1, 1024, decin1, 4096, 1024, 3072);
  gemm1(stream, decin1, decWp[0], Ybuf2, B, 512, 4096, 4096, 4096, 512);
  k_decnorm<<<1, 512, 0, stream>>>(Ybuf2, decG[0], decBe[0], decin2, 3072, 0, nullptr, B);
  // dec2 input: [fx1 | pool(concat[p1f2, cf2])]
  k_pool<<<(B * 640 + 255) / 256, 256, 0, stream>>>(f2[0], B, 128, 640, decin2, 3072, 512, 1792);
  k_pool<<<(B * 640 + 255) / 256, 256, 0, stream>>>(cf2,   B, 128, 640, decin2, 3072, 1152, 2432);
  gemm1(stream, decin2, decWp[1], Ybuf2, B, 512, 3072, 3072, 3072, 512);
  k_decnorm<<<1, 512, 0, stream>>>(Ybuf2, decG[1], decBe[1], decin3, 1792, 0, nullptr, B);
  // dec3 input: [fx2 | pool(concat[p1f1, cf1])]
  k_pool<<<(B * 320 + 255) / 256, 256, 0, stream>>>(f1[0], B, 512, 320, decin3, 1792, 512, 1152);
  k_pool<<<(B * 320 + 255) / 256, 256, 0, stream>>>(cf1,   B, 512, 320, decin3, 1792, 832, 1472);
  gemm1(stream, decin3, decWp[2], Ybuf2, B, 512, 1792, 1792, 1792, 512);
  k_decnorm<<<1, 512, 0, stream>>>(Ybuf2, decG[2], decBe[2], decdum, 512, 0, (float*)d_out, B);
}
